// PointerGeneratorNetwork_11871289606990
// MI455X (gfx1250) — matrix operand order repaired and checked by reading, not compiled, not run on hardware
//
#include <hip/hip_runtime.h>
#include <hip/hip_bf16.h>

// ---------------- problem constants ----------------
#define BB   8
#define UU   64
#define WMW  32
#define CAPN 1024
#define DECN 64
#define EE   128
#define HH   256
#define VV   32000
#define H4   1024   // 4*H
#define H2   512    // 2*H
#define H3   768    // 3*H

typedef _Float16 v16h __attribute__((ext_vector_type(16)));
typedef _Float16 v8h  __attribute__((ext_vector_type(8)));
typedef float    v8f  __attribute__((ext_vector_type(8)));

// ---------------- WMMA fragment loaders ----------------
// A: (M,K) row-major f16.  16-bit A 16x32 layout (contiguous k order, matching B):
//   lanes 0-15 : row M=lane, halves[0..15]=K(k0+g*16..k0+g*16+15), g = lane>>4
__device__ __forceinline__ v16h load_frag_a(const _Float16* A, int lda, int m0, int k0, int lane) {
    int r = lane & 15;
    int g = lane >> 4;
    const _Float16* base = A + (size_t)(m0 + r) * lda + k0;
    return *(const v16h*)(base + g * 16);
}

// B given as Bt: (N,K) row-major f16 (weight stored (out,in), computing x@W^T).
// B 32x16 contiguous-k layout: lane n (0-15) holds column N=n, halves = K(k0..k0+15); lanes 16-31: K(k0+16..k0+31).
__device__ __forceinline__ v16h load_frag_b(const _Float16* Bt, int ldb, int n0, int k0, int lane) {
    int c = lane & 15;
    int g = lane >> 4;
    return *(const v16h*)(Bt + (size_t)(n0 + c) * ldb + k0 + g * 16);
}

// ---------------- generic WMMA GEMM (16x64 per wave, register double-buffered) ----------------
// C(M,N) f32 = A(M,K) f16 * Bt(N,K)^T f16 (+ bias[n]); batched via grid.y.
// Each wave owns a 16x64 output strip: A fragment reused by 4 WMMAs per k-step;
// next k-step's fragments are issued before the current WMMAs so loads overlap matrix ops.
__global__ void gemm_wmma_kernel(const _Float16* __restrict__ A, int lda, long long sA,
                                 const _Float16* __restrict__ Bt, int ldb, long long sB,
                                 const float* __restrict__ bias,
                                 float* __restrict__ C, int ldc, long long sC,
                                 int M, int N, int K) {
    int batch = blockIdx.y;
    A  += (size_t)batch * sA;
    Bt += (size_t)batch * sB;
    C  += (size_t)batch * sC;

    int lane = threadIdx.x & 31;
    int wave = threadIdx.x >> 5;
    int ntiles = N >> 6;                       // 64-wide strips
    int tile = blockIdx.x * 8 + wave;
    int mt = tile / ntiles;
    int nt = tile - mt * ntiles;
    if (mt >= (M >> 4)) return;                // wave-uniform guard (EXEC all-ones for WMMA)
    int m0 = mt << 4, n0 = nt << 6;

    v8f acc0 = {}, acc1 = {}, acc2 = {}, acc3 = {};

    // prologue: k = 0 fragments
    v16h a  = load_frag_a(A, lda, m0, 0, lane);
    v16h b0 = load_frag_b(Bt, ldb, n0,      0, lane);
    v16h b1 = load_frag_b(Bt, ldb, n0 + 16, 0, lane);
    v16h b2 = load_frag_b(Bt, ldb, n0 + 32, 0, lane);
    v16h b3 = load_frag_b(Bt, ldb, n0 + 48, 0, lane);

    for (int k0 = 32; k0 < K; k0 += 32) {
        // issue next k-step loads first (keeps loads in flight under the WMMAs)
        __builtin_prefetch(A  + (size_t)(m0 + (lane & 15)) * lda + k0 + 32, 0, 3);
        __builtin_prefetch(Bt + (size_t)(n0 + (lane & 15)) * ldb + k0 + 32, 0, 3);
        v16h an  = load_frag_a(A, lda, m0, k0, lane);
        v16h b0n = load_frag_b(Bt, ldb, n0,      k0, lane);
        v16h b1n = load_frag_b(Bt, ldb, n0 + 16, k0, lane);
        v16h b2n = load_frag_b(Bt, ldb, n0 + 32, k0, lane);
        v16h b3n = load_frag_b(Bt, ldb, n0 + 48, k0, lane);
        acc0 = __builtin_amdgcn_wmma_f32_16x16x32_f16(false, a, false, b0, (short)0, acc0, false, false);
        acc1 = __builtin_amdgcn_wmma_f32_16x16x32_f16(false, a, false, b1, (short)0, acc1, false, false);
        acc2 = __builtin_amdgcn_wmma_f32_16x16x32_f16(false, a, false, b2, (short)0, acc2, false, false);
        acc3 = __builtin_amdgcn_wmma_f32_16x16x32_f16(false, a, false, b3, (short)0, acc3, false, false);
        a = an; b0 = b0n; b1 = b1n; b2 = b2n; b3 = b3n;
    }
    acc0 = __builtin_amdgcn_wmma_f32_16x16x32_f16(false, a, false, b0, (short)0, acc0, false, false);
    acc1 = __builtin_amdgcn_wmma_f32_16x16x32_f16(false, a, false, b1, (short)0, acc1, false, false);
    acc2 = __builtin_amdgcn_wmma_f32_16x16x32_f16(false, a, false, b2, (short)0, acc2, false, false);
    acc3 = __builtin_amdgcn_wmma_f32_16x16x32_f16(false, a, false, b3, (short)0, acc3, false, false);

    // epilogue: D layout -> row = m0 + (lane>=16 ? 8:0)+v, col = n0 + 16*j + (lane&15)
    int cl = lane & 15;
    int rbase = m0 + ((lane >> 4) << 3);
    v8f accs[4] = {acc0, acc1, acc2, acc3};
#pragma unroll
    for (int j = 0; j < 4; ++j) {
        int cc = n0 + j * 16 + cl;
        float bv = bias ? bias[cc] : 0.0f;
#pragma unroll
        for (int v = 0; v < 8; ++v) {
            int rr = rbase + v;
            if (rr < M) C[(size_t)rr * ldc + cc] = accs[j][v] + bv;
        }
    }
}

// ---------------- scatter: build input_pgn ----------------
__global__ void build_pgn_kernel(const int* __restrict__ input, const int* __restrict__ u_len,
                                 const int* __restrict__ w_len, int* __restrict__ pgn) {
    int b = blockIdx.x;
    for (int i = threadIdx.x; i < CAPN; i += blockDim.x) pgn[b * CAPN + i] = 0;
    __syncthreads();
    if (threadIdx.x == 0) {
        int csum = 0;
        int ul = u_len[b];
        for (int u = 0; u < UU; ++u) {
            int wl = w_len[b * UU + u];
            int start = csum;
            csum += wl;
            if ((csum <= CAPN) && (u < ul)) {
                for (int k = 0; k < wl; ++k)
                    pgn[b * CAPN + start + k] = input[(size_t)(b * UU + u) * WMW + k];
            }
        }
    }
}

// ---------------- embeddings ----------------
__global__ void embed_enc_kernel(const int* __restrict__ pgn, const float* __restrict__ emb,
                                 _Float16* __restrict__ xh) {
    int row = blockIdx.x;                  // B*CAP
    int c = threadIdx.x;                   // E
    int idx = pgn[row];
    float v = (idx != 0) ? emb[(size_t)idx * EE + c] : 0.0f;
    xh[(size_t)row * EE + c] = (_Float16)v;
}

__global__ void embed_dec_kernel(const int* __restrict__ target, const float* __restrict__ emb,
                                 _Float16* __restrict__ xh, float* __restrict__ xf) {
    int row = blockIdx.x;                  // B*DEC
    int c = threadIdx.x;                   // E
    int idx = target[row];
    float v = (idx != 0) ? emb[(size_t)idx * EE + c] : 0.0f;
    xh[(size_t)row * EE + c] = (_Float16)v;
    xf[(size_t)row * EE + c] = v;
}

// ---------------- f32 -> f16 convert ----------------
__global__ void f32_to_f16_kernel(const float* __restrict__ s, _Float16* __restrict__ d, int n) {
    int i = blockIdx.x * blockDim.x + threadIdx.x;
    if (i < n) d[i] = (_Float16)s[i];
}

// ---------------- persistent LSTM (sequential recurrence, WMMA for h@Whh^T) ----------------
__device__ __forceinline__ float sigf(float x) { return 1.0f / (1.0f + expf(-x)); }

// grid.x selects parameter set (enc fwd / enc bwd run concurrently in one launch).
// 1024 threads = 32 waves; wave w computes gate columns [w*32, w*32+32) sharing one A fragment.
__global__ void __launch_bounds__(1024)
lstm_seq_kernel(const float* __restrict__ xW_a, const _Float16* __restrict__ Whh_a,
                float* __restrict__ out_a, int rev_a,
                const float* __restrict__ xW_b, const _Float16* __restrict__ Whh_b,
                float* __restrict__ out_b, int rev_b,
                long long xWstrideB, int outLd, long long outStrideB, int T) {
    const float*    xW  = blockIdx.x ? xW_b  : xW_a;
    const _Float16* Whh = blockIdx.x ? Whh_b : Whh_a;
    float*          out = blockIdx.x ? out_b : out_a;
    int reverse         = blockIdx.x ? rev_b : rev_a;

    __shared__ _Float16 hsh[16 * HH];       // h state as f16 A-matrix, rows 8..15 = 0 pad
    __shared__ float    csh[BB * HH];       // cell state
    __shared__ float    gsh[BB * H4];       // gate preactivations

    int tid = threadIdx.x;
    int lane = tid & 31, wave = tid >> 5;

    for (int i = tid; i < 16 * HH; i += 1024) hsh[i] = (_Float16)0.0f;
    for (int i = tid; i < BB * HH; i += 1024) csh[i] = 0.0f;
    __syncthreads();

    for (int step = 0; step < T; ++step) {
        int t = reverse ? (T - 1 - step) : step;
        // --- g = h @ Whh^T (+ xW[t]) : K=256, two N-tiles per wave, shared A fragment ---
        int n0 = wave * 32;
        v8f acc0 = {}, acc1 = {};
#pragma unroll
        for (int k0 = 0; k0 < HH; k0 += 32) {
            v16h a  = load_frag_a(hsh, HH, 0, k0, lane);           // from LDS (ds_load)
            v16h b0 = load_frag_b(Whh, HH, n0,      k0, lane);     // from global (L2 resident)
            v16h b1 = load_frag_b(Whh, HH, n0 + 16, k0, lane);
            acc0 = __builtin_amdgcn_wmma_f32_16x16x32_f16(false, a, false, b0, (short)0, acc0, false, false);
            acc1 = __builtin_amdgcn_wmma_f32_16x16x32_f16(false, a, false, b1, (short)0, acc1, false, false);
        }
        {
            int cl = lane & 15;
            int rbase = (lane >> 4) << 3;
            v8f accs[2] = {acc0, acc1};
#pragma unroll
            for (int j = 0; j < 2; ++j) {
                int cc = n0 + j * 16 + cl;
#pragma unroll
                for (int v = 0; v < 8; ++v) {
                    int rr = rbase + v;
                    if (rr < BB)
                        gsh[rr * H4 + cc] = accs[j][v] + xW[(size_t)rr * xWstrideB + (size_t)t * H4 + cc];
                }
            }
        }
        __syncthreads();
        // --- gate nonlinearity + state update: 2048 (b,j) pairs over 1024 threads ---
#pragma unroll
        for (int rep = 0; rep < 2; ++rep) {
            int idx = tid + rep * 1024;
            int b = idx >> 8, j = idx & 255;
            float gi = gsh[b * H4 + j];
            float gf = gsh[b * H4 + HH + j];
            float gg = gsh[b * H4 + 2 * HH + j];
            float go = gsh[b * H4 + 3 * HH + j];
            float c = sigf(gf) * csh[b * HH + j] + sigf(gi) * tanhf(gg);
            float h = sigf(go) * tanhf(c);
            csh[b * HH + j] = c;
            hsh[b * HH + j] = (_Float16)h;
            out[(size_t)b * outStrideB + (size_t)t * outLd + j] = h;
        }
        __syncthreads();
    }
}

// ---------------- enc_out -> f16 + f16-transposed ----------------
__global__ void pack_enc_kernel(const float* __restrict__ eo, _Float16* __restrict__ eh,
                                _Float16* __restrict__ ehT) {
    int i = blockIdx.x * blockDim.x + threadIdx.x;          // B*CAP*2H
    if (i >= BB * CAPN * H2) return;
    int f = i & (H2 - 1);
    int rest = i >> 9;
    int e = rest & (CAPN - 1);
    int b = rest >> 10;
    _Float16 v = (_Float16)eo[i];
    eh[i] = v;
    ehT[((size_t)b * H2 + f) * CAPN + e] = v;
}

// ---------------- attention scores: et = v . tanh(Wh-part + Ws-part) ----------------
__global__ void attn_scores_kernel(const float* __restrict__ aWh, const float* __restrict__ aWs,
                                   const float* __restrict__ vvec, float* __restrict__ et) {
    int gw = blockIdx.x * 8 + (threadIdx.x >> 5);           // wave per (b,d,e)
    if (gw >= BB * DECN * CAPN) return;
    int e = gw & (CAPN - 1);
    int rest = gw >> 10;
    int d = rest & (DECN - 1);
    int b = rest >> 6;
    int lane = threadIdx.x & 31;
    const float* wh = aWh + ((size_t)b * CAPN + e) * HH;
    const float* ws = aWs + ((size_t)b * DECN + d) * HH;
    float s = 0.0f;
    for (int h = lane; h < HH; h += 32) s += vvec[h] * tanhf(wh[h] + ws[h]);
    for (int off = 16; off; off >>= 1) s += __shfl_xor(s, off, 32);
    if (lane == 0) et[gw] = s;
}

// ---------------- 256-thread block reduction helper ----------------
__device__ float block_reduce_256(float v, bool is_max) {
    __shared__ float tmp[9];
    int tid = threadIdx.x, lane = tid & 31, w = tid >> 5;
    __syncthreads();
    for (int off = 16; off; off >>= 1) {
        float o = __shfl_xor(v, off, 32);
        v = is_max ? fmaxf(v, o) : (v + o);
    }
    if (lane == 0) tmp[w] = v;
    __syncthreads();
    if (tid == 0) {
        float a = tmp[0];
        for (int i = 1; i < 8; ++i) a = is_max ? fmaxf(a, tmp[i]) : (a + tmp[i]);
        tmp[8] = a;
    }
    __syncthreads();
    return tmp[8];
}

// ---------------- attention softmax over CAP ----------------
__global__ void attn_softmax_kernel(const float* __restrict__ et, float* __restrict__ at,
                                    _Float16* __restrict__ ath) {
    int row = blockIdx.x;                                   // B*DEC
    const float* x = et + (size_t)row * CAPN;
    int tid = threadIdx.x;
    float m = -1e30f;
    for (int e = tid; e < CAPN; e += 256) m = fmaxf(m, x[e]);
    m = block_reduce_256(m, true);
    float s = 0.0f;
    for (int e = tid; e < CAPN; e += 256) s += expf(x[e] - m);
    s = block_reduce_256(s, false);
    float inv = 1.0f / s;
    for (int e = tid; e < CAPN; e += 256) {
        float a = expf(x[e] - m) * inv;
        at[(size_t)row * CAPN + e] = a;
        ath[(size_t)row * CAPN + e] = (_Float16)a;
    }
}

// ---------------- Pgen ----------------
__global__ void pgen_kernel(const float* __restrict__ ctx, const float* __restrict__ dec_out,
                            const float* __restrict__ dec_x, const float* __restrict__ wc,
                            const float* __restrict__ bc, const float* __restrict__ ws_p,
                            const float* __restrict__ wx, float* __restrict__ Pgen) {
    int r = blockIdx.x * 8 + (threadIdx.x >> 5);            // wave per row, 512 rows
    if (r >= BB * DECN) return;
    int lane = threadIdx.x & 31;
    float s = 0.0f;
    for (int f = lane; f < H2; f += 32) s += ctx[(size_t)r * H2 + f] * wc[f];
    for (int j = lane; j < HH; j += 32) s += dec_out[(size_t)r * HH + j] * ws_p[j];
    for (int k = lane; k < EE; k += 32) s += dec_x[(size_t)r * EE + k] * wx[k];
    for (int off = 16; off; off >>= 1) s += __shfl_xor(s, off, 32);
    if (lane == 0) Pgen[r] = 1.0f / (1.0f + expf(-(s + bc[0])));
}

// ---------------- [dec_out, ctx] concat to f16 ----------------
__global__ void pack_cat_kernel(const float* __restrict__ dec_out, const float* __restrict__ ctx,
                                _Float16* __restrict__ cath) {
    int i = blockIdx.x * blockDim.x + threadIdx.x;
    if (i >= BB * DECN * H3) return;
    int r = i / H3, c = i - r * H3;
    float v = (c < HH) ? dec_out[(size_t)r * HH + c] : ctx[(size_t)r * H2 + (c - HH)];
    cath[i] = (_Float16)v;
}

// ---------------- copy mass ----------------
__global__ void copymass_kernel(const int* __restrict__ pgn, const int* __restrict__ target,
                                const float* __restrict__ at, float* __restrict__ cm) {
    int gw = blockIdx.x * 8 + (threadIdx.x >> 5);           // wave per (b,d)
    if (gw >= BB * DECN) return;
    int b = gw >> 6, d = gw & (DECN - 1);
    int lane = threadIdx.x & 31;
    float s = 0.0f;
    if (d < DECN - 1) {
        int yt = target[b * DECN + d + 1];
        const int* p = pgn + (size_t)b * CAPN;
        const float* a = at + ((size_t)b * DECN + d) * CAPN;
        for (int e = lane; e < CAPN; e += 32) s += (p[e] == yt) ? a[e] : 0.0f;
    }
    for (int off = 16; off; off >>= 1) s += __shfl_xor(s, off, 32);
    if (lane == 0) cm[gw] = s;
}

// ---------------- fused final: softmax(logits) -> Pw -> log_softmax, in-place in d_out ----------------
__global__ void final_kernel(float* __restrict__ x_all, const float* __restrict__ Pgen,
                             const float* __restrict__ cm, const int* __restrict__ target) {
    int row = blockIdx.x;                                   // B*DEC
    int b = row >> 6, d = row & (DECN - 1);
    float* x = x_all + (size_t)row * VV;
    int tid = threadIdx.x;

    float m = -1e30f;
    for (int v = tid; v < VV; v += 256) m = fmaxf(m, x[v]);
    m = block_reduce_256(m, true);
    float z = 0.0f;
    for (int v = tid; v < VV; v += 256) z += expf(x[v] - m);
    z = block_reduce_256(z, false);
    float invz = 1.0f / z;

    float pg = Pgen[row];
    int ytv = (d < DECN - 1) ? target[b * DECN + d + 1] : -1;
    float corr = (d < DECN - 1) ? (1.0f - pg) * cm[row] : 0.0f;

    float s2 = 0.0f;
    for (int v = tid; v < VV; v += 256) {
        float pw = pg * expf(x[v] - m) * invz;
        if (v == ytv) pw += corr;                           // Ppt scatter (single index per row)
        x[v] = pw;
        s2 += expf(pw);                                     // Pw in [0,1] -> no max-shift needed
    }
    s2 = block_reduce_256(s2, false);
    float lz = logf(s2);
    for (int v = tid; v < VV; v += 256) x[v] = x[v] - lz;
}

// ---------------- host ----------------
extern "C" void kernel_launch(void* const* d_in, const int* in_sizes, int n_in,
                              void* d_out, int out_size, void* d_ws, size_t ws_size,
                              hipStream_t stream) {
    const int*   input   = (const int*)d_in[0];
    const int*   u_len   = (const int*)d_in[1];
    const int*   w_len   = (const int*)d_in[2];
    const int*   target  = (const int*)d_in[3];
    const float* enc_emb = (const float*)d_in[4];
    const float* dec_emb = (const float*)d_in[5];
    const float* Wih_ef  = (const float*)d_in[6];
    const float* Whh_ef  = (const float*)d_in[7];
    const float* b_ef    = (const float*)d_in[8];
    const float* Wih_eb  = (const float*)d_in[9];
    const float* Whh_eb  = (const float*)d_in[10];
    const float* b_eb    = (const float*)d_in[11];
    const float* Wih_d   = (const float*)d_in[12];
    const float* Whh_d   = (const float*)d_in[13];
    const float* b_d     = (const float*)d_in[14];
    const float* Wh      = (const float*)d_in[15];
    const float* Ws      = (const float*)d_in[16];
    const float* bs      = (const float*)d_in[17];
    const float* vvec    = (const float*)d_in[18];
    const float* W1      = (const float*)d_in[19];
    const float* b1      = (const float*)d_in[20];
    const float* W2      = (const float*)d_in[21];
    const float* b2      = (const float*)d_in[22];
    const float* wc      = (const float*)d_in[23];
    const float* bc      = (const float*)d_in[24];
    const float* ws_p    = (const float*)d_in[25];
    const float* wx      = (const float*)d_in[26];
    float* out = (float*)d_out;

    // ---- carve workspace (~142 MB total) ----
    char* p = (char*)d_ws;
    auto alloc = [&](size_t bytes) -> void* {
        void* r = (void*)p;
        p += (bytes + 255) & ~(size_t)255;
        return r;
    };
    int*       pgn       = (int*)alloc((size_t)BB * CAPN * 4);
    _Float16*  enc_xh    = (_Float16*)alloc((size_t)BB * CAPN * EE * 2);
    _Float16*  dec_xh    = (_Float16*)alloc((size_t)BB * DECN * EE * 2);
    float*     dec_x     = (float*)alloc((size_t)BB * DECN * EE * 4);
    _Float16*  Wih_ef_h  = (_Float16*)alloc((size_t)H4 * EE * 2);
    _Float16*  Whh_ef_h  = (_Float16*)alloc((size_t)H4 * HH * 2);
    _Float16*  Wih_eb_h  = (_Float16*)alloc((size_t)H4 * EE * 2);
    _Float16*  Whh_eb_h  = (_Float16*)alloc((size_t)H4 * HH * 2);
    _Float16*  Wih_d_h   = (_Float16*)alloc((size_t)H4 * EE * 2);
    _Float16*  Whh_d_h   = (_Float16*)alloc((size_t)H4 * HH * 2);
    _Float16*  Wh_h      = (_Float16*)alloc((size_t)HH * H2 * 2);
    _Float16*  Ws_h      = (_Float16*)alloc((size_t)HH * HH * 2);
    _Float16*  W1_h      = (_Float16*)alloc((size_t)HH * H3 * 2);
    _Float16*  W2_h      = (_Float16*)alloc((size_t)VV * HH * 2);
    float*     xW_ef     = (float*)alloc((size_t)BB * CAPN * H4 * 4);
    float*     xW_eb     = (float*)alloc((size_t)BB * CAPN * H4 * 4);
    float*     xW_d      = (float*)alloc((size_t)BB * DECN * H4 * 4);
    float*     enc_out   = (float*)alloc((size_t)BB * CAPN * H2 * 4);
    _Float16*  enc_out_h = (_Float16*)alloc((size_t)BB * CAPN * H2 * 2);
    _Float16*  enc_outT_h= (_Float16*)alloc((size_t)BB * H2 * CAPN * 2);
    float*     dec_out   = (float*)alloc((size_t)BB * DECN * HH * 4);
    _Float16*  dec_out_h = (_Float16*)alloc((size_t)BB * DECN * HH * 2);
    float*     attnWh    = (float*)alloc((size_t)BB * CAPN * HH * 4);
    float*     attnWs    = (float*)alloc((size_t)BB * DECN * HH * 4);
    float*     et        = (float*)alloc((size_t)BB * DECN * CAPN * 4);
    float*     at        = (float*)alloc((size_t)BB * DECN * CAPN * 4);
    _Float16*  at_h      = (_Float16*)alloc((size_t)BB * DECN * CAPN * 2);
    float*     ctx       = (float*)alloc((size_t)BB * DECN * H2 * 4);
    _Float16*  cat_h     = (_Float16*)alloc((size_t)BB * DECN * H3 * 2);
    float*     s1        = (float*)alloc((size_t)BB * DECN * HH * 4);
    _Float16*  s1_h      = (_Float16*)alloc((size_t)BB * DECN * HH * 2);
    float*     Pgen      = (float*)alloc((size_t)BB * DECN * 4);
    float*     cmass     = (float*)alloc((size_t)BB * DECN * 4);

    auto cvt = [&](const float* s, _Float16* d, int n) {
        f32_to_f16_kernel<<<(n + 255) / 256, 256, 0, stream>>>(s, d, n);
    };
    auto gemm = [&](const _Float16* A, int lda, long long sA,
                    const _Float16* Bt, int ldb, long long sB,
                    const float* bias, float* C, int ldc, long long sC,
                    int M, int N, int K, int batch) {
        int tiles = (M / 16) * (N / 64);       // 16x64 per wave
        dim3 grid((tiles + 7) / 8, batch);
        gemm_wmma_kernel<<<grid, 256, 0, stream>>>(A, lda, sA, Bt, ldb, sB, bias, C, ldc, sC, M, N, K);
    };

    // 1) weights -> f16
    cvt(Wih_ef, Wih_ef_h, H4 * EE);   cvt(Whh_ef, Whh_ef_h, H4 * HH);
    cvt(Wih_eb, Wih_eb_h, H4 * EE);   cvt(Whh_eb, Whh_eb_h, H4 * HH);
    cvt(Wih_d,  Wih_d_h,  H4 * EE);   cvt(Whh_d,  Whh_d_h,  H4 * HH);
    cvt(Wh, Wh_h, HH * H2);           cvt(Ws, Ws_h, HH * HH);
    cvt(W1, W1_h, HH * H3);           cvt(W2, W2_h, VV * HH);

    // 2) scatter + embeddings
    build_pgn_kernel<<<BB, 256, 0, stream>>>(input, u_len, w_len, pgn);
    embed_enc_kernel<<<BB * CAPN, EE, 0, stream>>>(pgn, enc_emb, enc_xh);
    embed_dec_kernel<<<BB * DECN, EE, 0, stream>>>(target, dec_emb, dec_xh, dec_x);

    // 3) input GEMMs (x @ Wih^T + b)
    gemm(enc_xh, EE, 0, Wih_ef_h, EE, 0, b_ef, xW_ef, H4, 0, BB * CAPN, H4, EE, 1);
    gemm(enc_xh, EE, 0, Wih_eb_h, EE, 0, b_eb, xW_eb, H4, 0, BB * CAPN, H4, EE, 1);
    gemm(dec_xh, EE, 0, Wih_d_h,  EE, 0, b_d,  xW_d,  H4, 0, BB * DECN, H4, EE, 1);

    // 4) LSTMs (enc fwd + bwd concurrently in one launch; dec separately)
    lstm_seq_kernel<<<2, 1024, 0, stream>>>(
        xW_ef, Whh_ef_h, enc_out, 0,
        xW_eb, Whh_eb_h, enc_out + HH, 1,
        (long long)CAPN * H4, H2, (long long)CAPN * H2, CAPN);
    lstm_seq_kernel<<<1, 1024, 0, stream>>>(
        xW_d, Whh_d_h, dec_out, 0,
        xW_d, Whh_d_h, dec_out, 0,
        (long long)DECN * H4, HH, (long long)DECN * HH, DECN);

    // 5) packs/converts for attention
    {
        int n = BB * CAPN * H2;
        pack_enc_kernel<<<(n + 255) / 256, 256, 0, stream>>>(enc_out, enc_out_h, enc_outT_h);
    }
    cvt(dec_out, dec_out_h, BB * DECN * HH);

    // 6) attention GEMMs
    gemm(enc_out_h, H2, 0, Wh_h, H2, 0, nullptr, attnWh, HH, 0, BB * CAPN, HH, H2, 1);
    gemm(dec_out_h, HH, 0, Ws_h, HH, 0, bs,      attnWs, HH, 0, BB * DECN, HH, HH, 1);

    // 7) scores + softmax
    attn_scores_kernel<<<(BB * DECN * CAPN) / 8, 256, 0, stream>>>(attnWh, attnWs, vvec, et);
    attn_softmax_kernel<<<BB * DECN, 256, 0, stream>>>(et, at, at_h);

    // 8) ctx = at @ enc_out   (batched over B; B operand = enc_out^T as (N=2H, K=CAP))
    gemm(at_h, CAPN, (long long)DECN * CAPN,
         enc_outT_h, CAPN, (long long)H2 * CAPN,
         nullptr, ctx, H2, (long long)DECN * H2,
         DECN, H2, CAPN, BB);

    // 9) Pgen
    pgen_kernel<<<(BB * DECN) / 8, 256, 0, stream>>>(ctx, dec_out, dec_x, wc, bc, ws_p, wx, Pgen);

    // 10) vocab projection: s1 = cat @ W1^T + b1 ; logits = s1 @ W2^T + b2 -> d_out
    {
        int n = BB * DECN * H3;
        pack_cat_kernel<<<(n + 255) / 256, 256, 0, stream>>>(dec_out, ctx, cat_h);
    }
    gemm(cat_h, H3, 0, W1_h, H3, 0, b1, s1, HH, 0, BB * DECN, HH, H3, 1);
    cvt(s1, s1_h, BB * DECN * HH);
    gemm(s1_h, HH, 0, W2_h, HH, 0, b2, out, VV, 0, BB * DECN, VV, HH, 1);

    // 11) pointer-copy mass, then fused softmax/mix/log-softmax in-place in d_out
    copymass_kernel<<<(BB * DECN) / 8, 256, 0, stream>>>(pgn, target, at, cmass);
    final_kernel<<<BB * DECN, 256, 0, stream>>>(out, Pgen, cmass, target);
}
